// MambaLayer2_CrossAttn_Click2_67001489817885
// MI455X (gfx1250) — matrix operand order repaired and checked by reading, not compiled, not run on hardware
//
#include <hip/hip_runtime.h>
#include <math.h>

#define DIMC 320
#define D_STATE 32
#define D_CONVK 4
#define HEADDIM 80
#define D_INNER 640
#define NHEADS 8
#define CONV_DIM 704
#define D_IN_PROJ 1352
#define AHEADS 4
#define MLP_DIMC 1024
#define INNERC 160
#define NPTS 6
#define BATCH 2
#define NTOK 16384
#define EPSV 1e-5f
#define GEMM_NT 4   // N-tiles (16 cols each) per wave

typedef __attribute__((ext_vector_type(16))) __bf16 v16bf;
typedef __attribute__((ext_vector_type(8)))  __bf16 v8bf;
typedef __attribute__((ext_vector_type(8)))  float  v8f;

__device__ __forceinline__ float sigmoidf_(float x){ return 1.f/(1.f+expf(-x)); }
__device__ __forceinline__ float siluf_(float x){ return x*sigmoidf_(x); }
__device__ __forceinline__ float softplusf_(float x){ return (x>20.f)? x : log1pf(expf(x)); }

// f32 -> bf16 elementwise conversion (staging for WMMA operands)
__global__ void cvt_bf16_kernel(const float* __restrict__ in, __bf16* __restrict__ out, long long n)
{
  long long i = (long long)blockIdx.x * blockDim.x + threadIdx.x;
  if (i < n) out[i] = (__bf16)in[i];
}

// ---------------------------------------------------------------------------
// GEMM: Y[M,N] = act( X[M,K] @ W[N,K]^T + bias + res )
// X, W already bf16 in memory: inner loop is pure b128 loads + WMMA, no VALU
// conversions. One wave per 16x(16*GEMM_NT) output strip; K % 32 == 0;
// out-of-range rows/cols clamped for loads (only unstored outputs affected).
// ---------------------------------------------------------------------------
__global__ void gemm_bf16_wmma(const __bf16* __restrict__ X, const __bf16* __restrict__ W,
                               const float* __restrict__ bias, const float* __restrict__ res,
                               float* __restrict__ Y, int M, int N, int K, int relu_act)
{
  int lane = threadIdx.x;
  int tm  = blockIdx.x * 16;
  int tn0 = blockIdx.y * (16 * GEMM_NT);
  int mrow = tm + (lane & 15);
  if (mrow >= M) mrow = M - 1;                 // clamp: affects only unstored rows
  int kofsA = (lane >> 4) << 4;                // A 16x32: lanes 16-31 -> K+16 (contiguous, matches B)
  int kofsB = (lane >> 4) << 4;                // B 32x16: lanes 16-31 -> K+16
  const __bf16* xrow = X + (size_t)mrow * K;
  const __bf16* wrow0;
  const __bf16* wrow1;
  const __bf16* wrow2;
  const __bf16* wrow3;
  {
    int n0 = tn0 + 0 * 16 + (lane & 15); wrow0 = W + (size_t)((n0 < N) ? n0 : (N - 1)) * K;
    int n1 = tn0 + 1 * 16 + (lane & 15); wrow1 = W + (size_t)((n1 < N) ? n1 : (N - 1)) * K;
    int n2 = tn0 + 2 * 16 + (lane & 15); wrow2 = W + (size_t)((n2 < N) ? n2 : (N - 1)) * K;
    int n3 = tn0 + 3 * 16 + (lane & 15); wrow3 = W + (size_t)((n3 < N) ? n3 : (N - 1)) * K;
  }
  v8f zero8 = {0.f,0.f,0.f,0.f,0.f,0.f,0.f,0.f};
  v8f acc0 = zero8, acc1 = zero8, acc2 = zero8, acc3 = zero8;

#pragma unroll 2
  for (int k0 = 0; k0 < K; k0 += 32) {
    if (k0 + 64 < K) {
      __builtin_prefetch(xrow + k0 + 64, 0, 1);
      __builtin_prefetch(wrow0 + k0 + 64, 0, 1);
    }
    // A fragment: two b128 loads of 8 bf16 each
    v8bf alo = *(const v8bf*)(xrow + k0 + kofsA);
    v8bf ahi = *(const v8bf*)(xrow + k0 + kofsA + 8);
    // B fragments: two b128 loads (16 contiguous bf16) per tile
    const v8bf* pb0 = (const v8bf*)(wrow0 + k0 + kofsB);
    const v8bf* pb1 = (const v8bf*)(wrow1 + k0 + kofsB);
    const v8bf* pb2 = (const v8bf*)(wrow2 + k0 + kofsB);
    const v8bf* pb3 = (const v8bf*)(wrow3 + k0 + kofsB);
    v8bf b0l = pb0[0], b0h = pb0[1];
    v8bf b1l = pb1[0], b1h = pb1[1];
    v8bf b2l = pb2[0], b2h = pb2[1];
    v8bf b3l = pb3[0], b3h = pb3[1];
    v16bf a  = __builtin_shufflevector(alo, ahi, 0,1,2,3,4,5,6,7,8,9,10,11,12,13,14,15);
    v16bf f0 = __builtin_shufflevector(b0l, b0h, 0,1,2,3,4,5,6,7,8,9,10,11,12,13,14,15);
    v16bf f1 = __builtin_shufflevector(b1l, b1h, 0,1,2,3,4,5,6,7,8,9,10,11,12,13,14,15);
    v16bf f2 = __builtin_shufflevector(b2l, b2h, 0,1,2,3,4,5,6,7,8,9,10,11,12,13,14,15);
    v16bf f3 = __builtin_shufflevector(b3l, b3h, 0,1,2,3,4,5,6,7,8,9,10,11,12,13,14,15);
    acc0 = __builtin_amdgcn_wmma_f32_16x16x32_bf16(false, a, false, f0, (short)0, acc0, false, false);
    acc1 = __builtin_amdgcn_wmma_f32_16x16x32_bf16(false, a, false, f1, (short)0, acc1, false, false);
    acc2 = __builtin_amdgcn_wmma_f32_16x16x32_bf16(false, a, false, f2, (short)0, acc2, false, false);
    acc3 = __builtin_amdgcn_wmma_f32_16x16x32_bf16(false, a, false, f3, (short)0, acc3, false, false);
  }

  v8f accs[GEMM_NT] = {acc0, acc1, acc2, acc3};
#pragma unroll
  for (int j = 0; j < GEMM_NT; ++j) {
    int on = tn0 + j * 16 + (lane & 15);
    if (on < N) {
      float bi = bias ? bias[on] : 0.f;
#pragma unroll
      for (int r = 0; r < 8; ++r) {
        int om = tm + r + ((lane >> 4) << 3);  // C/D: VGPR r -> M=r / M=r+8
        if (om < M) {
          float v = accs[j][r] + bi;
          if (res) v += res[(size_t)om * N + on];
          if (relu_act) v = fmaxf(v, 0.f);
          Y[(size_t)om * N + on] = v;
        }
      }
    }
  }
}

// ---------------------------------------------------------------------------
// LayerNorm of x (B,C,n) -> xln (B*n, C) row-major, one thread per token row
// ---------------------------------------------------------------------------
__global__ void lnx_kernel(const float* __restrict__ x, const float* __restrict__ w,
                           const float* __restrict__ b, float* __restrict__ out)
{
  int row = blockIdx.x * blockDim.x + threadIdx.x;
  if (row >= BATCH * NTOK) return;
  int bi = row / NTOK, t = row % NTOK;
  const float* xp = x + (size_t)bi * DIMC * NTOK + t;
  float s = 0.f;
  for (int c = 0; c < DIMC; ++c) s += xp[(size_t)c * NTOK];
  float m = s / DIMC;
  float v = 0.f;
  for (int c = 0; c < DIMC; ++c) { float d = xp[(size_t)c * NTOK] - m; v += d * d; }
  float inv = rsqrtf(v / DIMC + EPSV);
  float* op = out + (size_t)row * DIMC;
  for (int c = 0; c < DIMC; ++c) op[c] = (xp[(size_t)c * NTOK] - m) * inv * w[c] + b[c];
}

// causal depthwise conv over xBC slice of zxbcdt, + silu
__global__ void conv_silu_kernel(const float* __restrict__ zx, const float* __restrict__ cw,
                                 const float* __restrict__ cb, float* __restrict__ out)
{
  long long idx = (long long)blockIdx.x * blockDim.x + threadIdx.x;
  long long total = (long long)BATCH * NTOK * CONV_DIM;
  if (idx >= total) return;
  int c = (int)(idx % CONV_DIM);
  int t = (int)((idx / CONV_DIM) % NTOK);
  int b = (int)(idx / ((long long)CONV_DIM * NTOK));
  float acc = cb[c];
#pragma unroll
  for (int k = 0; k < D_CONVK; ++k) {
    int tt = t + k - (D_CONVK - 1);
    if (tt >= 0) acc += zx[((size_t)(b * NTOK + tt)) * D_IN_PROJ + D_INNER + c] * cw[c * D_CONVK + k];
  }
  out[idx] = siluf_(acc);
}

// ---------------------------------------------------------------------------
// SSM scan: one block per (b,h), state S[32,80] in registers (8 per thread)
// blockDim = (80,4): x=p, y=n-group of 8
// ---------------------------------------------------------------------------
__global__ void ssm_scan_kernel(const float* __restrict__ zx, const float* __restrict__ xbc,
                                const float* __restrict__ dt_bias, const float* __restrict__ A_log,
                                const float* __restrict__ Dp, float* __restrict__ ybuf)
{
  int b = blockIdx.x / NHEADS;
  int h = blockIdx.x % NHEADS;
  int tx = threadIdx.x;            // 0..79  (p)
  int ty = threadIdx.y;            // 0..3   (n group)
  int tid = ty * 80 + tx;
  __shared__ float xsh[HEADDIM], Bsh[D_STATE], Csh[D_STATE];
  __shared__ float dtsh, decsh;
  __shared__ float ypart[4][HEADDIM];
  float Ah = -expf(A_log[h]);
  float Dh = Dp[h];
  float dtb = dt_bias[h];
  float S[8];
#pragma unroll
  for (int i = 0; i < 8; ++i) S[i] = 0.f;
  for (int t = 0; t < NTOK; ++t) {
    size_t rowc = ((size_t)(b * NTOK + t)) * CONV_DIM;
    if (tid < HEADDIM) xsh[tid] = xbc[rowc + h * HEADDIM + tid];
    else if (tid < HEADDIM + D_STATE) Bsh[tid - HEADDIM] = xbc[rowc + D_INNER + (tid - HEADDIM)];
    else if (tid < HEADDIM + 2 * D_STATE) Csh[tid - HEADDIM - D_STATE] = xbc[rowc + D_INNER + D_STATE + (tid - HEADDIM - D_STATE)];
    else if (tid == HEADDIM + 2 * D_STATE) {
      float dtv = softplusf_(zx[((size_t)(b * NTOK + t)) * D_IN_PROJ + (D_INNER + CONV_DIM) + h] + dtb);
      dtsh = dtv;
      decsh = expf(dtv * Ah);
    }
    __syncthreads();
    float xv = xsh[tx];
    float dtv = dtsh, dec = decsh;
    float yp = 0.f;
#pragma unroll
    for (int i = 0; i < 8; ++i) {
      int n = ty * 8 + i;
      float s = S[i] * dec + (dtv * Bsh[n]) * xv;
      S[i] = s;
      yp += Csh[n] * s;
    }
    ypart[ty][tx] = yp;
    __syncthreads();
    if (ty == 0) {
      float y = ypart[0][tx] + ypart[1][tx] + ypart[2][tx] + ypart[3][tx] + Dh * xv;
      ybuf[((size_t)(b * NTOK + t)) * D_INNER + h * HEADDIM + tx] = y;
    }
  }
}

// y = y*silu(z); y = y * rsqrt(mean(y^2)+eps) * rms_w  (in place), one block per row
__global__ void gate_rms_kernel(float* __restrict__ y, const float* __restrict__ zx,
                                const float* __restrict__ rms_w)
{
  int row = blockIdx.x;
  int tid = threadIdx.x;
  __shared__ float red[256];
  float vv[3];
  float ss = 0.f;
  int cnt = 0;
  for (int c = tid; c < D_INNER; c += 256) {
    float z = zx[(size_t)row * D_IN_PROJ + c];
    float val = y[(size_t)row * D_INNER + c] * siluf_(z);
    vv[cnt++] = val;
    ss += val * val;
  }
  red[tid] = ss; __syncthreads();
  for (int st = 128; st > 0; st >>= 1) { if (tid < st) red[tid] += red[tid + st]; __syncthreads(); }
  float inv = rsqrtf(red[0] / D_INNER + EPSV);
  cnt = 0;
  for (int c = tid; c < D_INNER; c += 256)
    y[(size_t)row * D_INNER + c] = vv[cnt++] * inv * rms_w[c];
}

// dense grid positional encoding: kpe[n,320]
__global__ void kpe_kernel(const float* __restrict__ G, float* __restrict__ kpe)
{
  int idx = blockIdx.x * blockDim.x + threadIdx.x;
  if (idx >= NTOK * 160) return;
  int j = idx % 160;
  int t = idx / 160;
  int d = t >> 10, hh = (t >> 5) & 31, w = t & 31;
  float gd = 2.f * ((d + 0.5f) / 16.f) - 1.f;
  float gh = 2.f * ((hh + 0.5f) / 32.f) - 1.f;
  float gw = 2.f * ((w + 0.5f) / 32.f) - 1.f;
  float c = 6.283185307179586f * (gd * G[j] + gh * G[160 + j] + gw * G[320 + j]);
  kpe[(size_t)t * DIMC + j] = sinf(c);
  kpe[(size_t)t * DIMC + 160 + j] = cosf(c);
}

// point embeddings -> qpe and queries
__global__ void point_emb_kernel(const float* __restrict__ coords, const int* __restrict__ labels,
                                 const float* __restrict__ G, const float* __restrict__ ptab,
                                 float* __restrict__ qpe, float* __restrict__ queries)
{
  int idx = blockIdx.x * blockDim.x + threadIdx.x;
  if (idx >= BATCH * NPTS * 160) return;
  int j = idx % 160;
  int p = (idx / 160) % NPTS;
  int b = idx / (160 * NPTS);
  const float* cc = coords + (b * NPTS + p) * 3;
  float g0 = 2.f * (cc[0] / 128.f) - 1.f;
  float g1 = 2.f * (cc[1] / 256.f) - 1.f;
  float g2 = 2.f * (cc[2] / 256.f) - 1.f;
  float c = 6.283185307179586f * (g0 * G[j] + g1 * G[160 + j] + g2 * G[320 + j]);
  int lab = labels[b * NPTS + p];
  float s  = sinf(c) + ptab[lab * DIMC + j];
  float co = cosf(c) + ptab[lab * DIMC + 160 + j];
  size_t base = (size_t)(b * NPTS + p) * DIMC;
  qpe[base + j] = s;       qpe[base + 160 + j] = co;
  queries[base + j] = s;   queries[base + 160 + j] = co;
}

__global__ void add_bcast_kernel(const float* __restrict__ a, const float* __restrict__ b,
                                 float* __restrict__ o, long long total, long long bmod)
{
  long long i = (long long)blockIdx.x * blockDim.x + threadIdx.x;
  if (i < total) o[i] = a[i] + b[i % bmod];
}

// out = LN(A (+ Badd)) * w + bias ; one block per row, C <= 1024
__global__ void add_ln_kernel(const float* __restrict__ A, const float* __restrict__ Badd,
                              const float* __restrict__ w, const float* __restrict__ bias,
                              float* __restrict__ out, int C)
{
  int row = blockIdx.x;
  int tid = threadIdx.x;
  __shared__ float red[256];
  float vv[4];
  int cnt = 0;
  float s = 0.f;
  for (int c = tid; c < C; c += 256) {
    float v = A[(size_t)row * C + c];
    if (Badd) v += Badd[(size_t)row * C + c];
    vv[cnt++] = v;
    s += v;
  }
  red[tid] = s; __syncthreads();
  for (int st = 128; st > 0; st >>= 1) { if (tid < st) red[tid] += red[tid + st]; __syncthreads(); }
  float m = red[0] / C;
  __syncthreads();
  float v2 = 0.f;
  cnt = 0;
  for (int c = tid; c < C; c += 256) { float d = vv[cnt++] - m; v2 += d * d; }
  red[tid] = v2; __syncthreads();
  for (int st = 128; st > 0; st >>= 1) { if (tid < st) red[tid] += red[tid + st]; __syncthreads(); }
  float inv = rsqrtf(red[0] / C + EPSV);
  cnt = 0;
  for (int c = tid; c < C; c += 256)
    out[(size_t)row * C + c] = (vv[cnt++] - m) * inv * w[c] + bias[c];
}

// tiny 6-token self attention, hd=80, heads=4; block per (b,h), thread per query
__global__ void self_attn_kernel(const float* __restrict__ q, const float* __restrict__ k,
                                 const float* __restrict__ v, float* __restrict__ out)
{
  int b = blockIdx.x / AHEADS;
  int h = blockIdx.x % AHEADS;
  int qi = threadIdx.x;
  if (qi >= NPTS) return;
  const float* qr = q + (size_t)(b * NPTS + qi) * DIMC + h * 80;
  float sc[NPTS];
  float mx = -1e30f;
  for (int kk = 0; kk < NPTS; ++kk) {
    const float* kr = k + (size_t)(b * NPTS + kk) * DIMC + h * 80;
    float s = 0.f;
    for (int j = 0; j < 80; ++j) s += qr[j] * kr[j];
    s *= 0.11180339887498949f;   // 1/sqrt(80)
    sc[kk] = s; mx = fmaxf(mx, s);
  }
  float sum = 0.f;
  for (int kk = 0; kk < NPTS; ++kk) { sc[kk] = expf(sc[kk] - mx); sum += sc[kk]; }
  float inv = 1.f / sum;
  float* orow = out + (size_t)(b * NPTS + qi) * DIMC + h * 80;
  for (int j = 0; j < 80; ++j) {
    float acc = 0.f;
    for (int kk = 0; kk < NPTS; ++kk) acc += sc[kk] * v[(size_t)(b * NPTS + kk) * DIMC + h * 80 + j];
    orow[j] = acc * inv;
  }
}

// t2i attention: 6 queries over 16384 keys, hd=40; block per (b,h,qi), online softmax
__global__ void t2i_attn_kernel(const float* __restrict__ q, const float* __restrict__ k,
                                const float* __restrict__ v, float* __restrict__ out)
{
  int blk = blockIdx.x;
  int b  = blk / (AHEADS * NPTS);
  int h  = (blk / NPTS) % AHEADS;
  int qi = blk % NPTS;
  int tid = threadIdx.x;
  __shared__ float qsh[40];
  __shared__ float sm[256], ssum[256];
  __shared__ float sacc[256 * 40];
  if (tid < 40) qsh[tid] = q[(size_t)(b * NPTS + qi) * INNERC + h * 40 + tid];
  __syncthreads();
  const float rscale = 0.15811388300841897f;   // 1/sqrt(40)
  float m = -1e30f, s = 0.f;
  float acc[40];
#pragma unroll
  for (int j = 0; j < 40; ++j) acc[j] = 0.f;
  for (int t = tid; t < NTOK; t += 256) {
    const float* kr = k + ((size_t)(b * NTOK + t)) * INNERC + h * 40;
    float sc = 0.f;
#pragma unroll
    for (int j = 0; j < 40; ++j) sc += qsh[j] * kr[j];
    sc *= rscale;
    if (sc > m) {
      float f = expf(m - sc);
      s *= f;
#pragma unroll
      for (int j = 0; j < 40; ++j) acc[j] *= f;
      m = sc;
    }
    float e = expf(sc - m);
    s += e;
    const float* vr = v + ((size_t)(b * NTOK + t)) * INNERC + h * 40;
#pragma unroll
    for (int j = 0; j < 40; ++j) acc[j] += e * vr[j];
  }
  sm[tid] = m; __syncthreads();
  for (int st = 128; st > 0; st >>= 1) { if (tid < st) sm[tid] = fmaxf(sm[tid], sm[tid + st]); __syncthreads(); }
  float gm = sm[0];
  float f = expf(m - gm);
  ssum[tid] = s * f;
#pragma unroll
  for (int j = 0; j < 40; ++j) sacc[tid * 40 + j] = acc[j] * f;
  __syncthreads();
  for (int st = 128; st > 0; st >>= 1) {
    if (tid < st) {
      ssum[tid] += ssum[tid + st];
      for (int j = 0; j < 40; ++j) sacc[tid * 40 + j] += sacc[(tid + st) * 40 + j];
    }
    __syncthreads();
  }
  if (tid < 40)
    out[(size_t)(b * NPTS + qi) * INNERC + h * 40 + tid] = sacc[tid] / ssum[0];
}

// i2t attention: 16384 queries over 6 keys, hd=40; thread per (b,t)
__global__ void i2t_attn_kernel(const float* __restrict__ q, const float* __restrict__ k,
                                const float* __restrict__ v, float* __restrict__ out)
{
  int idx = blockIdx.x * blockDim.x + threadIdx.x;
  if (idx >= BATCH * NTOK) return;
  int b = idx / NTOK;
  const float rscale = 0.15811388300841897f;
  for (int h = 0; h < AHEADS; ++h) {
    const float* qr = q + (size_t)idx * INNERC + h * 40;
    float sc[NPTS];
    float mx = -1e30f;
    for (int kk = 0; kk < NPTS; ++kk) {
      const float* kr = k + (size_t)(b * NPTS + kk) * INNERC + h * 40;
      float s = 0.f;
      for (int j = 0; j < 40; ++j) s += qr[j] * kr[j];
      s *= rscale;
      sc[kk] = s; mx = fmaxf(mx, s);
    }
    float sum = 0.f;
    for (int kk = 0; kk < NPTS; ++kk) { sc[kk] = expf(sc[kk] - mx); sum += sc[kk]; }
    float inv = 1.f / sum;
    for (int j = 0; j < 40; ++j) {
      float a = 0.f;
      for (int kk = 0; kk < NPTS; ++kk) a += sc[kk] * v[(size_t)(b * NPTS + kk) * INNERC + h * 40 + j];
      out[(size_t)idx * INNERC + h * 40 + j] = a * inv;
    }
  }
}

__global__ void transpose_out_kernel(const float* __restrict__ keys, float* __restrict__ out)
{
  long long idx = (long long)blockIdx.x * blockDim.x + threadIdx.x;
  long long total = (long long)BATCH * DIMC * NTOK;
  if (idx >= total) return;
  int t = (int)(idx % NTOK);
  int c = (int)((idx / NTOK) % DIMC);
  int b = (int)(idx / ((long long)NTOK * DIMC));
  out[idx] = keys[(size_t)(b * NTOK + t) * DIMC + c];
}

// ---------------------------------------------------------------------------
extern "C" void kernel_launch(void* const* d_in, const int* in_sizes, int n_in,
                              void* d_out, int out_size, void* d_ws, size_t ws_size,
                              hipStream_t stream)
{
  (void)in_sizes; (void)n_in; (void)out_size; (void)ws_size;
  const float* x        = (const float*)d_in[0];
  const float* coords   = (const float*)d_in[1];
  const int*   labels   = (const int*)d_in[2];
  const float* ln_w     = (const float*)d_in[3];
  const float* ln_b     = (const float*)d_in[4];
  const float* in_w     = (const float*)d_in[5];
  const float* conv_w   = (const float*)d_in[6];
  const float* conv_b   = (const float*)d_in[7];
  const float* dt_bias  = (const float*)d_in[8];
  const float* A_log    = (const float*)d_in[9];
  const float* Dp       = (const float*)d_in[10];
  const float* rms_w    = (const float*)d_in[11];
  const float* out_w    = (const float*)d_in[12];
  const float* pe_gauss = (const float*)d_in[13];
  const float* ptab     = (const float*)d_in[14];
  const float* sa_w     = (const float*)d_in[15];
  const float* sa_b     = (const float*)d_in[16];
  const float* t2i_w    = (const float*)d_in[17];
  const float* t2i_b    = (const float*)d_in[18];
  const float* t2i_ow   = (const float*)d_in[19];
  const float* t2i_ob   = (const float*)d_in[20];
  const float* i2t_w    = (const float*)d_in[21];
  const float* i2t_b    = (const float*)d_in[22];
  const float* i2t_ow   = (const float*)d_in[23];
  const float* i2t_ob   = (const float*)d_in[24];
  const float* norms_w  = (const float*)d_in[25];
  const float* norms_b  = (const float*)d_in[26];
  const float* mlp_w1   = (const float*)d_in[27];
  const float* mlp_b1   = (const float*)d_in[28];
  const float* mlp_w2   = (const float*)d_in[29];
  const float* mlp_b2   = (const float*)d_in[30];
  float* d_outf = (float*)d_out;

  float* ws = (float*)d_ws;
  const size_t ROWS = (size_t)BATCH * NTOK;     // 32768
  size_t o = 0;
  float* f_xln   = ws + o; o += ROWS * DIMC;        // later aliased as kplus
  float* f_zx    = ws + o; o += ROWS * D_IN_PROJ;   // later aliased as i2t tmp320
  float* f_conv  = ws + o; o += ROWS * CONV_DIM;
  float* f_y     = ws + o; o += ROWS * D_INNER;
  float* f_keys  = ws + o; o += ROWS * DIMC;
  float* f_kpe   = ws + o; o += (size_t)NTOK * DIMC;
  float* f_kproj = ws + o; o += ROWS * INNERC;
  float* f_vproj = ws + o; o += ROWS * INNERC;
  float* f_qpe   = ws + o; o += 12 * DIMC;
  float* f_qry   = ws + o; o += 12 * DIMC;
  float* f_qplus = ws + o; o += 12 * DIMC;
  float* f_pq    = ws + o; o += 12 * DIMC;
  float* f_pk    = ws + o; o += 12 * DIMC;
  float* f_pv    = ws + o; o += 12 * DIMC;
  float* f_atts  = ws + o; o += 12 * DIMC;
  float* f_mlph  = ws + o; o += 12 * MLP_DIMC;
  float* f_tmpq  = ws + o; o += 12 * DIMC;

  // bf16 arena (16B aligned: o is a multiple of 4 floats)
  __bf16* bf = (__bf16*)(ws + o);
  size_t bo = 0;
  __bf16* bw_in   = bf + bo; bo += (size_t)D_IN_PROJ * DIMC;
  __bf16* bw_out  = bf + bo; bo += (size_t)DIMC * D_INNER;
  __bf16* bw_sa   = bf + bo; bo += (size_t)2 * 4 * DIMC * DIMC;
  __bf16* bw_t2i  = bf + bo; bo += (size_t)2 * 3 * INNERC * DIMC;
  __bf16* bw_t2io = bf + bo; bo += (size_t)2 * DIMC * INNERC;
  __bf16* bw_i2t  = bf + bo; bo += (size_t)2 * 3 * INNERC * DIMC;
  __bf16* bw_i2to = bf + bo; bo += (size_t)2 * DIMC * INNERC;
  __bf16* bw_mlp1 = bf + bo; bo += (size_t)2 * MLP_DIMC * DIMC;
  __bf16* bw_mlp2 = bf + bo; bo += (size_t)2 * DIMC * MLP_DIMC;
  __bf16* bx      = bf + bo; bo += ROWS * D_INNER;   // activation staging (max M*K)

  auto cvt = [&](const float* src, __bf16* dst, long long n) {
    cvt_bf16_kernel<<<(int)((n + 255) / 256), 256, 0, stream>>>(src, dst, n);
  };
  const unsigned NSTRIP = 16 * GEMM_NT;  // 64 output columns per wave
  auto rungemm = [&](const float* Xf, const __bf16* Wb, const float* bias, const float* res,
                     float* Y, int M, int N, int K, int relu) {
    cvt(Xf, bx, (long long)M * K);
    dim3 g((unsigned)((M + 15) / 16), (N + NSTRIP - 1) / NSTRIP);
    gemm_bf16_wmma<<<g, 32, 0, stream>>>(bx, Wb, bias, res, Y, M, N, K, relu);
  };

  // convert all GEMM weights to bf16 once per launch
  cvt(in_w,   bw_in,   (long long)D_IN_PROJ * DIMC);
  cvt(out_w,  bw_out,  (long long)DIMC * D_INNER);
  cvt(sa_w,   bw_sa,   (long long)2 * 4 * DIMC * DIMC);
  cvt(t2i_w,  bw_t2i,  (long long)2 * 3 * INNERC * DIMC);
  cvt(t2i_ow, bw_t2io, (long long)2 * DIMC * INNERC);
  cvt(i2t_w,  bw_i2t,  (long long)2 * 3 * INNERC * DIMC);
  cvt(i2t_ow, bw_i2to, (long long)2 * DIMC * INNERC);
  cvt(mlp_w1, bw_mlp1, (long long)2 * MLP_DIMC * DIMC);
  cvt(mlp_w2, bw_mlp2, (long long)2 * DIMC * MLP_DIMC);

  // 1. LN over channels of x, transposed to token-major
  lnx_kernel<<<(int)((ROWS + 255) / 256), 256, 0, stream>>>(x, ln_w, ln_b, f_xln);
  // 2. in_proj GEMM (32768x320 @ 1352x320^T)
  rungemm(f_xln, bw_in, nullptr, nullptr, f_zx, (int)ROWS, D_IN_PROJ, DIMC, 0);
  // 3. causal depthwise conv + silu
  { long long tot = (long long)ROWS * CONV_DIM;
    conv_silu_kernel<<<(int)((tot + 255) / 256), 256, 0, stream>>>(f_zx, conv_w, conv_b, f_conv); }
  // 4. sequential SSM scan (16 blocks = B*H)
  ssm_scan_kernel<<<BATCH * NHEADS, dim3(80, 4), 0, stream>>>(f_zx, f_conv, dt_bias, A_log, Dp, f_y);
  // 5. gate with silu(z) + RMSNorm (in place)
  gate_rms_kernel<<<(int)ROWS, 256, 0, stream>>>(f_y, f_zx, rms_w);
  // 6. out_proj -> keys
  rungemm(f_y, bw_out, nullptr, nullptr, f_keys, (int)ROWS, DIMC, D_INNER, 0);
  // 7. dense grid PE
  kpe_kernel<<<(NTOK * 160 + 255) / 256, 256, 0, stream>>>(pe_gauss, f_kpe);
  // 8. point embeddings -> qpe and queries
  point_emb_kernel<<<(BATCH * NPTS * 160 + 255) / 256, 256, 0, stream>>>(coords, labels, pe_gauss, ptab, f_qpe, f_qry);

  const int QR = BATCH * NPTS;  // 12
  for (int i = 0; i < 2; ++i) {
    const __bf16* saw = bw_sa + (size_t)i * 4 * DIMC * DIMC;
    const float*  sab = sa_b + (size_t)i * 4 * DIMC;
    // --- self attention (skip PE on first layer) ---
    const float* qin = f_qry;
    if (i != 0) {
      add_bcast_kernel<<<(QR * DIMC + 255) / 256, 256, 0, stream>>>(f_qry, f_qpe, f_qplus, QR * DIMC, QR * DIMC);
      qin = f_qplus;
    }
    rungemm(qin, saw + 0 * DIMC * DIMC, sab + 0 * DIMC, nullptr, f_pq, QR, DIMC, DIMC, 0);
    rungemm(qin, saw + 1 * DIMC * DIMC, sab + 1 * DIMC, nullptr, f_pk, QR, DIMC, DIMC, 0);
    rungemm(f_qry, saw + 2 * DIMC * DIMC, sab + 2 * DIMC, nullptr, f_pv, QR, DIMC, DIMC, 0);
    self_attn_kernel<<<BATCH * AHEADS, 32, 0, stream>>>(f_pq, f_pk, f_pv, f_atts);
    rungemm(f_atts, saw + 3 * DIMC * DIMC, sab + 3 * DIMC,
            (i == 0) ? nullptr : f_qry, f_tmpq, QR, DIMC, DIMC, 0);
    add_ln_kernel<<<QR, 256, 0, stream>>>(f_tmpq, nullptr, norms_w + (i * 4 + 0) * DIMC, norms_b + (i * 4 + 0) * DIMC, f_qry, DIMC);
    // --- t2i cross attention ---
    add_bcast_kernel<<<(QR * DIMC + 255) / 256, 256, 0, stream>>>(f_qry, f_qpe, f_qplus, QR * DIMC, QR * DIMC);
    float* f_kplus = f_xln;  // reuse
    { long long tot = (long long)ROWS * DIMC;
      add_bcast_kernel<<<(int)((tot + 255) / 256), 256, 0, stream>>>(f_keys, f_kpe, f_kplus, tot, (long long)NTOK * DIMC); }
    const __bf16* tw = bw_t2i + (size_t)i * 3 * INNERC * DIMC;
    const float*  tb = t2i_b + (size_t)i * 3 * INNERC;
    rungemm(f_qplus, tw + 0 * INNERC * DIMC, tb + 0 * INNERC, nullptr, f_pq, QR, INNERC, DIMC, 0);
    rungemm(f_kplus, tw + 1 * INNERC * DIMC, tb + 1 * INNERC, nullptr, f_kproj, (int)ROWS, INNERC, DIMC, 0);
    rungemm(f_keys, tw + 2 * INNERC * DIMC, tb + 2 * INNERC, nullptr, f_vproj, (int)ROWS, INNERC, DIMC, 0);
    t2i_attn_kernel<<<BATCH * AHEADS * NPTS, 256, 0, stream>>>(f_pq, f_kproj, f_vproj, f_atts);
    rungemm(f_atts, bw_t2io + (size_t)i * DIMC * INNERC, t2i_ob + i * DIMC, f_qry, f_tmpq, QR, DIMC, INNERC, 0);
    add_ln_kernel<<<QR, 256, 0, stream>>>(f_tmpq, nullptr, norms_w + (i * 4 + 1) * DIMC, norms_b + (i * 4 + 1) * DIMC, f_qry, DIMC);
    // --- MLP ---
    rungemm(f_qry, bw_mlp1 + (size_t)i * MLP_DIMC * DIMC, mlp_b1 + i * MLP_DIMC, nullptr, f_mlph, QR, MLP_DIMC, DIMC, 1);
    rungemm(f_mlph, bw_mlp2 + (size_t)i * DIMC * MLP_DIMC, mlp_b2 + i * DIMC, f_qry, f_tmpq, QR, DIMC, MLP_DIMC, 0);
    add_ln_kernel<<<QR, 256, 0, stream>>>(f_tmpq, nullptr, norms_w + (i * 4 + 2) * DIMC, norms_b + (i * 4 + 2) * DIMC, f_qry, DIMC);
    // --- i2t cross attention (keys attend to queries) ---
    add_bcast_kernel<<<(QR * DIMC + 255) / 256, 256, 0, stream>>>(f_qry, f_qpe, f_qplus, QR * DIMC, QR * DIMC);
    const __bf16* iw = bw_i2t + (size_t)i * 3 * INNERC * DIMC;
    const float*  ib = i2t_b + (size_t)i * 3 * INNERC;
    rungemm(f_kplus, iw + 0 * INNERC * DIMC, ib + 0 * INNERC, nullptr, f_kproj, (int)ROWS, INNERC, DIMC, 0);
    rungemm(f_qplus, iw + 1 * INNERC * DIMC, ib + 1 * INNERC, nullptr, f_pk, QR, INNERC, DIMC, 0);
    rungemm(f_qry, iw + 2 * INNERC * DIMC, ib + 2 * INNERC, nullptr, f_pv, QR, INNERC, DIMC, 0);
    i2t_attn_kernel<<<(int)((ROWS + 255) / 256), 256, 0, stream>>>(f_kproj, f_pk, f_pv, f_vproj);
    float* f_tmp320 = f_zx;  // reuse
    rungemm(f_vproj, bw_i2to + (size_t)i * DIMC * INNERC, i2t_ob + i * DIMC, nullptr, f_tmp320, (int)ROWS, DIMC, INNERC, 0);
    add_ln_kernel<<<(int)ROWS, 256, 0, stream>>>(f_keys, f_tmp320, norms_w + (i * 4 + 3) * DIMC, norms_b + (i * 4 + 3) * DIMC, f_keys, DIMC);
  }
  // final transpose (B,n,C) -> (B,C,d,h,w)
  transpose_out_kernel<<<(int)(((long long)BATCH * DIMC * NTOK + 255) / 256), 256, 0, stream>>>(f_keys, d_outf);
}
